// HandwritingGenerator_73409581023587
// MI455X (gfx1250) — matrix operand order repaired and checked by reading, not compiled, not run on hardware
//
#include <hip/hip_runtime.h>
#include <hip/hip_bf16.h>

// ---------------- model constants ----------------
constexpr int Bx = 64, Tx = 600, Ux = 64, Vx = 80, Hx = 512, KAx = 10, KOx = 20;
constexpr int BT = Bx * Tx;             // 38400
constexpr int Gx = 4 * Hx;              // 2048 gates
constexpr int K0 = 608;                 // lstm0 fused K: 96 (pad of 83) + 512
constexpr int KX = 608;                 // lstm1/2 input K: 595 padded to 608
constexpr int KF = 1536;                // fc K
constexpr int NF = 128;                 // fc N padded (121 real)

typedef __attribute__((ext_vector_type(16))) _Float16 v16h;
typedef __attribute__((ext_vector_type(8)))  float    v8f;

static __device__ __forceinline__ v8f vzero8() {
  v8f z = {0.f, 0.f, 0.f, 0.f, 0.f, 0.f, 0.f, 0.f};
  return z;
}
static __device__ __forceinline__ float sigm(float x) { return 1.f / (1.f + expf(-x)); }

// A fragment (16x32 f16, M x K): lane l holds row M=l&15; K halves {0..7,16..23} (l<16)
// or {8..15,24..31} (l>=16), per ISA 7.12.2 "16-bit A-Matrix 16x32".
static __device__ __forceinline__ v16h ldfragA(const _Float16* base, int ld, int kk, int lane) {
  const _Float16* p = base + (lane & 15) * ld + kk * 32 + ((lane < 16) ? 0 : 8);
  union { uint4 u[2]; v16h v; } r;
  r.u[0] = *reinterpret_cast<const uint4*>(p);
  r.u[1] = *reinterpret_cast<const uint4*>(p + 16);
  return r.v;
}
// B fragment (32x16 f16, K x N), weights stored row-major (N,K): lane l holds col N=l&15,
// K halves {0..7,16..23} (l<16) or {8..15,24..31} (l>=16), matching A.
static __device__ __forceinline__ v16h ldfragB(const _Float16* rowp, int kk, int lane) {
  const _Float16* p = rowp + kk * 32 + ((lane < 16) ? 0 : 8);
  union { uint4 u[2]; v16h v; } r;
  r.u[0] = *reinterpret_cast<const uint4*>(p);
  r.u[1] = *reinterpret_cast<const uint4*>(p + 16);
  return r.v;
}
static __device__ __forceinline__ v8f wmma32(v16h a, v16h b, v8f c) {
  return __builtin_amdgcn_wmma_f32_16x16x32_f16(false, a, false, b, (short)0, c, false, false);
}

// ---------------- prep kernels ----------------
__global__ void k_cast_pad(_Float16* dst, const float* src, int rows, int sk, int dk) {
  int i = blockIdx.x * 256 + threadIdx.x;
  if (i >= rows * dk) return;
  int r = i / dk, c = i % dk;
  dst[i] = (_Float16)(c < sk ? src[r * sk + c] : 0.f);
}
__global__ void k_cast_pad_rows(_Float16* dst, const float* src, int drows, int srows, int k) {
  int i = blockIdx.x * 256 + threadIdx.x;
  if (i >= drows * k) return;
  int r = i / k, c = i % k;
  dst[i] = (_Float16)(r < srows ? src[r * k + c] : 0.f);
}
// Wcat0[n, 0..82]=lstm0_Wih, [83..95]=0, [96..607]=lstm0_Whh
__global__ void k_pack_w0(_Float16* dst, const float* wih, const float* whh) {
  int i = blockIdx.x * 256 + threadIdx.x;
  if (i >= Gx * K0) return;
  int r = i / K0, c = i % K0;
  float v = (c < 83) ? wih[r * 83 + c] : ((c >= 96) ? whh[r * 512 + (c - 96)] : 0.f);
  dst[i] = (_Float16)v;
}
__global__ void k_char_oh(float* dst, const int* seq, const float* mask) {
  int i = blockIdx.x * 256 + threadIdx.x;
  if (i >= Bx * Ux * Vx) return;
  int bu = i / Vx, v = i % Vx;
  dst[i] = (seq[bu] == v) ? mask[bu] : 0.f;
}
// xin rows: cols 0..2 strokes, cols 595..607 zero pad
__global__ void k_xin_static(_Float16* xin, const float* strokes) {
  int i = blockIdx.x * 256 + threadIdx.x;
  if (i >= BT * 16) return;
  int row = i / 16, c = i % 16;
  if (c < 3) xin[row * KX + c] = (_Float16)strokes[row * 3 + c];
  else       xin[row * KX + 592 + c] = (_Float16)0.f;   // 595..607
}

// Software-pipelined K-loop body: preload kk+1 fragments before issuing kk's WMMAs.
#define WMMA_PIPELINED_LOOP(ABASE, ALD, KST)                                  \
  {                                                                           \
    v16h a  = ldfragA(ABASE, ALD, 0, lane);                                   \
    v16h b0 = ldfragB(w0, 0, lane), b1 = ldfragB(w1, 0, lane);                \
    v16h b2 = ldfragB(w2, 0, lane), b3 = ldfragB(w3, 0, lane);                \
    for (int kk = 0; kk < (KST); ++kk) {                                      \
      int kn = (kk + 1 < (KST)) ? kk + 1 : kk;                                \
      v16h an  = ldfragA(ABASE, ALD, kn, lane);                               \
      v16h nb0 = ldfragB(w0, kn, lane), nb1 = ldfragB(w1, kn, lane);          \
      v16h nb2 = ldfragB(w2, kn, lane), nb3 = ldfragB(w3, kn, lane);          \
      ac0 = wmma32(a, b0, ac0);                                               \
      ac1 = wmma32(a, b1, ac1);                                               \
      ac2 = wmma32(a, b2, ac2);                                               \
      ac3 = wmma32(a, b3, ac3);                                               \
      a = an; b0 = nb0; b1 = nb1; b2 = nb2; b3 = nb3;                         \
    }                                                                         \
  }

// ---------------- persistent lstm0 + attention ----------------
__global__ __launch_bounds__(1024) void k_lstm0_attn(
    const _Float16* __restrict__ Wcat, const float* __restrict__ bias,
    const float* __restrict__ strokes, const float* __restrict__ winW,
    const float* __restrict__ winb, const float* __restrict__ cmask,
    const float* __restrict__ choh,
    _Float16* __restrict__ hcat, _Float16* __restrict__ xin1, _Float16* __restrict__ xin2) {
  extern __shared__ char smem[];
  _Float16* sA  = (_Float16*)smem;                       // 64 x 608 f16
  float*    sP  = (float*)(smem + Bx * K0 * 2);          // 64 x 32
  float*    sKp = sP + Bx * 32;                          // 64 x 16
  float*    sPh = sKp + Bx * 16;                         // 64 x 64
  float*    sW  = sPh + Bx * Ux;                         // 64 x 80
  const int tid = threadIdx.x, lane = tid & 31, wv = tid >> 5;

  for (int i = tid; i < Bx * K0; i += 1024) sA[i] = (_Float16)0.f;
  for (int i = tid; i < Bx * 16; i += 1024) sKp[i] = 0.f;
  for (int i = tid; i < Bx * Vx; i += 1024) sW[i] = 0.f;
  __syncthreads();

  float cst[4][8], hrg[4][8];
  for (int q = 0; q < 4; ++q)
    for (int r = 0; r < 8; ++r) cst[q][r] = 0.f;

  for (int t = 0; t < Tx; ++t) {
    // phase 1: inputs [x_t | w_prev] into A
    for (int i = tid; i < Bx * Vx; i += 1024) {
      int b = i / Vx, v = i % Vx;
      sA[b * K0 + 3 + v] = (_Float16)sW[i];
    }
    if (tid < Bx * 3) {
      int b = tid / 3, j = tid % 3;
      sA[b * K0 + j] = (_Float16)strokes[(b * Tx + t) * 3 + j];
    }
    __syncthreads();

    // phase 2: gates = [x|w|h] @ Wcat^T, then LSTM cell (state in VGPRs)
    for (int q = 0; q < 4; ++q) {
      int g = wv * 4 + q;
      int mb = (g >> 5) * 16, nb = (g & 31) * 16;
      int col = nb + (lane & 15);
      v8f ac0 = vzero8(), ac1 = vzero8(), ac2 = vzero8(), ac3 = vzero8();
      const _Float16* abase = sA + mb * K0;
      const _Float16* w0 = Wcat + (size_t)(col) * K0;
      const _Float16* w1 = Wcat + (size_t)(512 + col) * K0;
      const _Float16* w2 = Wcat + (size_t)(1024 + col) * K0;
      const _Float16* w3 = Wcat + (size_t)(1536 + col) * K0;
      WMMA_PIPELINED_LOOP(abase, K0, K0 / 32)
      float bi = bias[col], bf = bias[512 + col], bg = bias[1024 + col], bo = bias[1536 + col];
      for (int r = 0; r < 8; ++r) {
        float iv = ac0[r] + bi, fv = ac1[r] + bf, gv = ac2[r] + bg, ov = ac3[r] + bo;
        float c = sigm(fv) * cst[q][r] + sigm(iv) * tanhf(gv);
        cst[q][r] = c;
        hrg[q][r] = sigm(ov) * tanhf(c);
      }
    }
    __syncthreads();   // all waves done reading h_{t-1}
    // phase 3: publish h_t
    for (int q = 0; q < 4; ++q) {
      int g = wv * 4 + q;
      int mb = (g >> 5) * 16, nb = (g & 31) * 16;
      int col = nb + (lane & 15);
      int rbase = mb + ((lane >> 4) * 8);
      for (int r = 0; r < 8; ++r) {
        int b = rbase + r;
        _Float16 hv = (_Float16)hrg[q][r];
        sA[b * K0 + 96 + col] = hv;
        int row = b * Tx + t;
        hcat[row * KF + col] = hv;
        xin1[row * KX + 83 + col] = hv;
      }
    }
    __syncthreads();
    // phase 4: p = h @ winW^T + winb
    for (int i = tid; i < Bx * 30; i += 1024) {
      int b = i / 30, j = i % 30;
      float s = winb[j];
      const _Float16* hp = sA + b * K0 + 96;
      const float* wp = winW + j * Hx;
      for (int k2 = 0; k2 < Hx; ++k2) s += (float)hp[k2] * wp[k2];
      sP[b * 32 + j] = s;
    }
    __syncthreads();
    // phase 5: kappa += exp(kap_hat)
    if (tid < Bx * KAx) {
      int b = tid / KAx, a2 = tid % KAx;
      sKp[b * 16 + a2] += expf(sP[b * 32 + 20 + a2]);
    }
    __syncthreads();
    // phase 6: phi
    for (int i = tid; i < Bx * Ux; i += 1024) {
      int b = i >> 6, u = i & 63;
      float s = 0.f;
      for (int a2 = 0; a2 < KAx; ++a2) {
        float alpha = expf(sP[b * 32 + a2]);
        float beta  = expf(sP[b * 32 + 10 + a2]);
        float d = sKp[b * 16 + a2] - (float)u;
        s += alpha * expf(-beta * d * d);
      }
      sPh[i] = s * cmask[i];
    }
    __syncthreads();
    // phase 7: w = phi @ char_oh
    for (int i = tid; i < Bx * Vx; i += 1024) {
      int b = i / Vx, v = i % Vx;
      float s = 0.f;
      const float* ph = sPh + b * Ux;
      const float* oh = choh + (size_t)(b * Ux) * Vx + v;
      for (int u = 0; u < Ux; ++u) s += ph[u] * oh[u * Vx];
      sW[i] = s;
      _Float16 w16 = (_Float16)s;
      int row = b * Tx + t;
      xin1[row * KX + 3 + v] = w16;
      xin2[row * KX + 3 + v] = w16;
    }
    __syncthreads();
  }
}

// ---------------- persistent recurrent LSTM (gx precomputed) ----------------
__global__ __launch_bounds__(1024) void k_lstm_rec(
    const float* __restrict__ gx, const _Float16* __restrict__ Whh,
    _Float16* __restrict__ hcat, int hcol, _Float16* __restrict__ xin_next) {
  extern __shared__ char smem[];
  _Float16* sH = (_Float16*)smem;   // 64 x 512 f16
  const int tid = threadIdx.x, lane = tid & 31, wv = tid >> 5;
  for (int i = tid; i < Bx * Hx; i += 1024) sH[i] = (_Float16)0.f;
  __syncthreads();

  float cst[4][8], hrg[4][8];
  for (int q = 0; q < 4; ++q)
    for (int r = 0; r < 8; ++r) cst[q][r] = 0.f;

  for (int t = 0; t < Tx; ++t) {
    for (int q = 0; q < 4; ++q) {
      int g = wv * 4 + q;
      int mb = (g >> 5) * 16, nb = (g & 31) * 16;
      int col = nb + (lane & 15);
      int rbase = mb + ((lane >> 4) * 8);
      v8f ac0 = vzero8(), ac1 = vzero8(), ac2 = vzero8(), ac3 = vzero8();
      for (int r = 0; r < 8; ++r) {
        const float* gp = gx + (size_t)((rbase + r) * Tx + t) * Gx + col;
        ac0[r] = gp[0]; ac1[r] = gp[512]; ac2[r] = gp[1024]; ac3[r] = gp[1536];
      }
      const _Float16* abase = sH + mb * Hx;
      const _Float16* w0 = Whh + (size_t)(col) * Hx;
      const _Float16* w1 = Whh + (size_t)(512 + col) * Hx;
      const _Float16* w2 = Whh + (size_t)(1024 + col) * Hx;
      const _Float16* w3 = Whh + (size_t)(1536 + col) * Hx;
      WMMA_PIPELINED_LOOP(abase, Hx, Hx / 32)
      for (int r = 0; r < 8; ++r) {
        float c = sigm(ac1[r]) * cst[q][r] + sigm(ac0[r]) * tanhf(ac2[r]);
        cst[q][r] = c;
        hrg[q][r] = sigm(ac3[r]) * tanhf(c);
      }
    }
    __syncthreads();
    for (int q = 0; q < 4; ++q) {
      int g = wv * 4 + q;
      int mb = (g >> 5) * 16, nb = (g & 31) * 16;
      int col = nb + (lane & 15);
      int rbase = mb + ((lane >> 4) * 8);
      for (int r = 0; r < 8; ++r) {
        int b = rbase + r;
        _Float16 hv = (_Float16)hrg[q][r];
        sH[b * Hx + col] = hv;
        int row = b * Tx + t;
        hcat[row * KF + hcol + col] = hv;
        if (xin_next) xin_next[row * KX + 83 + col] = hv;
      }
    }
    __syncthreads();
  }
}

// ---------------- big parallel WMMA GEMM: C = A @ W^T (+bias) ----------------
// Block = 8 waves, all sharing one 16-row A tile staged in LDS; each wave owns a
// 64-column group. Grid.x = Mt * ceil(Ngrp/8) (a block never straddles an M tile).
__global__ __launch_bounds__(256) void k_gemm(const _Float16* __restrict__ A, int lda,
                                              const _Float16* __restrict__ W, int ldw,
                                              float* __restrict__ C, int ldc,
                                              int Ngrp, int Kst,
                                              const float* __restrict__ bias, int nbias) {
  extern __shared__ char smem[];
  _Float16* sAt = (_Float16*)smem;   // 16 x K
  const int K = Kst * 32;
  const int ngBlocks = (Ngrp + 7) / 8;
  const int mi = blockIdx.x / ngBlocks;
  const int ngb = blockIdx.x % ngBlocks;
  const int wv = threadIdx.x >> 5, lane = threadIdx.x & 31;

  // cooperative A-tile staging (16B chunks)
  const _Float16* ag = A + (size_t)(mi * 16) * lda;
  const int chunksPerRow = K / 8;
  for (int i = threadIdx.x; i < 16 * chunksPerRow; i += 256) {
    int row = i / chunksPerRow, c8 = i % chunksPerRow;
    *reinterpret_cast<uint4*>(sAt + row * K + c8 * 8) =
        *reinterpret_cast<const uint4*>(ag + (size_t)row * lda + c8 * 8);
  }
  __syncthreads();

  const int ng = ngb * 8 + wv;
  if (ng >= Ngrp) return;   // wave-uniform exit (EXEC all ones inside)
  const int ncol = ng * 64 + (lane & 15);
  const _Float16* w0 = W + (size_t)(ncol) * ldw;
  const _Float16* w1 = W + (size_t)(ncol + 16) * ldw;
  const _Float16* w2 = W + (size_t)(ncol + 32) * ldw;
  const _Float16* w3 = W + (size_t)(ncol + 48) * ldw;
  v8f ac0 = vzero8(), ac1 = vzero8(), ac2 = vzero8(), ac3 = vzero8();
  WMMA_PIPELINED_LOOP(sAt, K, Kst)

  const int rbase = mi * 16 + ((lane >> 4) * 8);
  const int c0 = ncol, c1 = ncol + 16, c2 = ncol + 32, c3 = ncol + 48;
  const float b0 = (bias && c0 < nbias) ? bias[c0] : 0.f;
  const float b1 = (bias && c1 < nbias) ? bias[c1] : 0.f;
  const float b2 = (bias && c2 < nbias) ? bias[c2] : 0.f;
  const float b3 = (bias && c3 < nbias) ? bias[c3] : 0.f;
  for (int r = 0; r < 8; ++r) {
    float* cp = C + (size_t)(rbase + r) * ldc;
    cp[c0] = ac0[r] + b0;
    cp[c1] = ac1[r] + b1;
    cp[c2] = ac2[r] + b2;
    cp[c3] = ac3[r] + b3;
  }
}

// ---------------- MDN post-processing into tuple layout ----------------
__global__ void k_post(const float* __restrict__ F, float* __restrict__ out) {
  int bt = blockIdx.x * blockDim.x + threadIdx.x;
  if (bt >= BT) return;
  const float* f = F + (size_t)bt * NF;
  float mx = f[80];
  for (int j = 1; j < KOx; ++j) mx = fmaxf(mx, f[80 + j]);
  float s = 0.f;
  for (int j = 0; j < KOx; ++j) s += expf(f[80 + j] - mx);
  float lse = mx + logf(s);
  float* lpi = out;                     // BT*20
  float* myu = out + (size_t)BT * 20;   // BT*40
  float* lsg = out + (size_t)BT * 60;   // BT*40
  float* rho = out + (size_t)BT * 100;  // BT*20
  float* eos = out + (size_t)BT * 120;  // BT*1
  for (int j = 0; j < KOx; ++j) lpi[(size_t)bt * 20 + j] = f[80 + j] - lse;
  for (int m = 0; m < 40; ++m) myu[(size_t)bt * 40 + m] = f[m];
  for (int m = 0; m < 40; ++m) lsg[(size_t)bt * 40 + m] = f[40 + m];
  for (int j = 0; j < KOx; ++j) rho[(size_t)bt * 20 + j] = tanhf(f[100 + j]);
  eos[bt] = 1.f / (1.f + expf(f[120]));   // sigmoid(-eos)
}

// ---------------- host ----------------
extern "C" void kernel_launch(void* const* d_in, const int* in_sizes, int n_in,
                              void* d_out, int out_size, void* d_ws, size_t ws_size,
                              hipStream_t stream) {
  const int*   char_seq  = (const int*)d_in[0];
  const float* char_mask = (const float*)d_in[1];
  const float* strokes   = (const float*)d_in[2];
  const float* l0Wih = (const float*)d_in[4];
  const float* l0Whh = (const float*)d_in[5];
  const float* l0b   = (const float*)d_in[6];
  const float* winW  = (const float*)d_in[7];
  const float* winb  = (const float*)d_in[8];
  const float* l1Wih = (const float*)d_in[9];
  const float* l1Whh = (const float*)d_in[10];
  const float* l1b   = (const float*)d_in[11];
  const float* l2Wih = (const float*)d_in[12];
  const float* l2Whh = (const float*)d_in[13];
  const float* l2b   = (const float*)d_in[14];
  const float* fcW   = (const float*)d_in[15];
  const float* fcb   = (const float*)d_in[16];

  size_t cur = 0;
  auto carve = [&](size_t bytes) -> char* {
    char* p = (char*)d_ws + cur;
    cur += (bytes + 255) & ~(size_t)255;
    return p;
  };
  _Float16* hcat  = (_Float16*)carve((size_t)BT * KF * 2);
  _Float16* xin1  = (_Float16*)carve((size_t)BT * KX * 2);
  _Float16* xin2  = (_Float16*)carve((size_t)BT * KX * 2);
  float*    gx    = (float*)   carve((size_t)BT * Gx * 4);
  float*    fcout = (float*)   carve((size_t)BT * NF * 4);
  _Float16* W0c   = (_Float16*)carve((size_t)Gx * K0 * 2);
  _Float16* W1i   = (_Float16*)carve((size_t)Gx * KX * 2);
  _Float16* W1h   = (_Float16*)carve((size_t)Gx * Hx * 2);
  _Float16* W2i   = (_Float16*)carve((size_t)Gx * KX * 2);
  _Float16* W2h   = (_Float16*)carve((size_t)Gx * Hx * 2);
  _Float16* fcWp  = (_Float16*)carve((size_t)NF * KF * 2);
  float*    choh  = (float*)   carve((size_t)Bx * Ux * Vx * 4);

  auto nb = [](int n) { return (n + 255) / 256; };
  k_pack_w0<<<nb(Gx * K0), 256, 0, stream>>>(W0c, l0Wih, l0Whh);
  k_cast_pad<<<nb(Gx * KX), 256, 0, stream>>>(W1i, l1Wih, Gx, 595, KX);
  k_cast_pad<<<nb(Gx * Hx), 256, 0, stream>>>(W1h, l1Whh, Gx, Hx, Hx);
  k_cast_pad<<<nb(Gx * KX), 256, 0, stream>>>(W2i, l2Wih, Gx, 595, KX);
  k_cast_pad<<<nb(Gx * Hx), 256, 0, stream>>>(W2h, l2Whh, Gx, Hx, Hx);
  k_cast_pad_rows<<<nb(NF * KF), 256, 0, stream>>>(fcWp, fcW, NF, 121, KF);
  k_char_oh<<<nb(Bx * Ux * Vx), 256, 0, stream>>>(choh, char_seq, char_mask);
  k_xin_static<<<nb(BT * 16), 256, 0, stream>>>(xin1, strokes);
  k_xin_static<<<nb(BT * 16), 256, 0, stream>>>(xin2, strokes);

  // lstm0 + attention (sequential, persistent single WGP workgroup)
  size_t sh0 = (size_t)Bx * K0 * 2 + (size_t)Bx * (32 + 16 + Ux + Vx) * 4;
  k_lstm0_attn<<<1, 1024, sh0, stream>>>(W0c, l0b, strokes, winW, winb,
                                         char_mask, choh, hcat, xin1, xin2);

  // lstm1: gx = xin1 @ W1ih^T + b ; then recurrent
  int Mt = BT / 16;                        // 2400
  int gridG = Mt * ((Gx / 64 + 7) / 8);    // 2400 * 4
  size_t shG = (size_t)16 * KX * 2;
  k_gemm<<<gridG, 256, shG, stream>>>(xin1, KX, W1i, KX, gx, Gx, Gx / 64, KX / 32, l1b, Gx);
  k_lstm_rec<<<1, 1024, (size_t)Bx * Hx * 2, stream>>>(gx, W1h, hcat, 512, xin2);

  // lstm2
  k_gemm<<<gridG, 256, shG, stream>>>(xin2, KX, W2i, KX, gx, Gx, Gx / 64, KX / 32, l2b, Gx);
  k_lstm_rec<<<1, 1024, (size_t)Bx * Hx * 2, stream>>>(gx, W2h, hcat, 1024, nullptr);

  // fc head
  int gridF = Mt * ((NF / 64 + 7) / 8);    // 2400
  size_t shF = (size_t)16 * KF * 2;
  k_gemm<<<gridF, 256, shF, stream>>>(hcat, KF, fcWp, KF, fcout, NF, NF / 64, KF / 32, fcb, 121);
  k_post<<<nb(BT), 256, 0, stream>>>(fcout, (float*)d_out);
}
